// MultiHeadAttention_78967268704380
// MI455X (gfx1250) — compile-verified, matrix operand order checked by reading, not run on hardware
//
#include <hip/hip_runtime.h>

typedef __bf16 bf16;
typedef __attribute__((ext_vector_type(2)))  __bf16 v2bf;
typedef __attribute__((ext_vector_type(8)))  __bf16 v8bf;
typedef __attribute__((ext_vector_type(16))) __bf16 v16bf;
typedef __attribute__((ext_vector_type(8)))  float  v8f;

#define D_MODEL 1024
#define N_HEADS 16
#define DK      64
#define B_      4
#define T_      2048
#define QKV_N   (3 * D_MODEL)   // 3072
#define ROWS    (B_ * T_)       // 8192

// ---------------------------------------------------------------------------
// Fragment loader for V_WMMA_F32_16X16X32_BF16 (16-bit A 16x32 layout; B is
// fed as transposed-A). lane 0-15 -> row=lane, K in {0..7,16..23};
// lane 16-31 -> row=lane-16, K in {8..15,24..31}. Two 16B chunks per lane.
// ---------------------------------------------------------------------------
__device__ __forceinline__ v16bf load_frag(const bf16* base, int stride, int lane) {
    const int row = lane & 15;
    const int kb  = (lane >> 4) << 3;
    const bf16* p = base + row * stride + kb;
    v8bf lo = *(const v8bf*)(p);
    v8bf hi = *(const v8bf*)(p + 16);
    return __builtin_shufflevector(lo, hi, 0,1,2,3,4,5,6,7,8,9,10,11,12,13,14,15);
}

__device__ __forceinline__ v8f wmma_bf16(v16bf a, v16bf b, v8f c) {
    return __builtin_amdgcn_wmma_f32_16x16x32_bf16(false, a, false, b, (short)0, c,
                                                   false, false);
}

// ---------------------------------------------------------------------------
// Async global->LDS copy (CDNA5, ASYNCcnt-tracked). One 16B line per lane.
// ---------------------------------------------------------------------------
__device__ __forceinline__ void async_copy_b128(void* lds_ptr, const void* gptr) {
    unsigned lds_off = (unsigned)(size_t)lds_ptr;   // low 32 bits = LDS offset
    asm volatile("global_load_async_to_lds_b128 %0, %1, off"
                 :: "v"(lds_off), "v"(gptr) : "memory");
}
__device__ __forceinline__ void async_wait0() {
    asm volatile("s_wait_asynccnt 0" ::: "memory");
}

// ---------------------------------------------------------------------------
// Raw hardware transcendentals (v_exp_f32 is base-2; no libm range fixup).
// ---------------------------------------------------------------------------
__device__ __forceinline__ float fast_exp2(float x) { return __builtin_amdgcn_exp2f(x); }
__device__ __forceinline__ float fast_rcp(float x)  { return __builtin_amdgcn_rcpf(x); }

// ---------------------------------------------------------------------------
// 16-lane (half-wave) reductions on the VALU pipe via DPP (no LDS traffic).
// ---------------------------------------------------------------------------
template <int M>
__device__ __forceinline__ float dpp8_xor(float x) {
    constexpr unsigned sel = (unsigned)((0^M)      | ((1^M)<<3)  | ((2^M)<<6)  |
                                        ((3^M)<<9) | ((4^M)<<12) | ((5^M)<<15) |
                                        ((6^M)<<18)| ((7^M)<<21));
    int r = __builtin_amdgcn_mov_dpp8(__builtin_bit_cast(int, x), sel);
    return __builtin_bit_cast(float, r);
}
__device__ __forceinline__ float dpp_row_mirror(float x) {
    int r = __builtin_amdgcn_update_dpp(__builtin_bit_cast(int, x),
                                        __builtin_bit_cast(int, x),
                                        0x140 /*ROW_MIRROR*/, 0xf, 0xf, true);
    return __builtin_bit_cast(float, r);
}
__device__ __forceinline__ float half_reduce_max(float v) {
    v = fmaxf(v, dpp8_xor<1>(v));
    v = fmaxf(v, dpp8_xor<2>(v));
    v = fmaxf(v, dpp8_xor<4>(v));
    v = fmaxf(v, dpp_row_mirror(v));
    return v;
}
__device__ __forceinline__ float half_reduce_sum(float v) {
    v += dpp8_xor<1>(v);
    v += dpp8_xor<2>(v);
    v += dpp8_xor<4>(v);
    v += dpp_row_mirror(v);
    return v;
}

// ---------------------------------------------------------------------------
// fp32 -> bf16 streaming convert
// ---------------------------------------------------------------------------
__global__ __launch_bounds__(256) void cvt_f32_bf16(const float* __restrict__ in,
                                                    bf16* __restrict__ out, int n4) {
    int i = blockIdx.x * blockDim.x + threadIdx.x;
    const int stride = gridDim.x * blockDim.x;
    for (; i < n4; i += stride) {
        float4 v = ((const float4*)in)[i];
        bf16* p = out + (size_t)i * 4;
        p[0] = (bf16)v.x; p[1] = (bf16)v.y; p[2] = (bf16)v.z; p[3] = (bf16)v.w;
    }
}

// ---------------------------------------------------------------------------
// Tiled bf16 GEMM:  C[M,N] = A[M,K] * B[K,N] + bias[N]
// 128 threads = 4 waves. 64(M) x 128(N) tile, K stepped by 64, double-buffered:
// async A-tile + global B-tile loads for step s+1 are issued before the WMMAs
// of step s; the register->LDS transpose stores land after the WMMAs.
// ---------------------------------------------------------------------------
template <bool BF16OUT>
__global__ __launch_bounds__(128) void gemm_bf16_kernel(
    const bf16* __restrict__ A, const bf16* __restrict__ Bm,
    const float* __restrict__ bias, void* __restrict__ Cout,
    int M, int N, int K) {
    __shared__ __align__(16) bf16 As[2][64 * 72];    // 64 rows x 64 k
    __shared__ __align__(16) bf16 Bts[2][128 * 72];  // 128 n-rows x 64 k

    const int t    = threadIdx.x;
    const int lane = t & 31;
    const int w    = t >> 5;
    const int n0   = blockIdx.x * 128;
    const int m0   = blockIdx.y * 64;

    v8f acc[8] = {};

    auto issueA = [&](int buf, int k0) {
#pragma unroll
        for (int i = 0; i < 4; ++i) {
            int c = t + i * 128;
            int row = c >> 3, coff = (c & 7) * 8;
            async_copy_b128(&As[buf][row * 72 + coff],
                            &A[(size_t)(m0 + row) * K + k0 + coff]);
        }
    };
    v8bf br0[4], br1[4];
    auto loadB = [&](int k0) {
#pragma unroll
        for (int i = 0; i < 4; ++i) {
            int p = t + i * 128;
            int kp = p >> 4, nc = (p & 15) * 8;
            int k = 2 * kp;
            br0[i] = *(const v8bf*)&Bm[(size_t)(k0 + k)     * N + n0 + nc];
            br1[i] = *(const v8bf*)&Bm[(size_t)(k0 + k + 1) * N + n0 + nc];
        }
    };
    auto storeB = [&](int buf) {
#pragma unroll
        for (int i = 0; i < 4; ++i) {
            int p = t + i * 128;
            int kp = p >> 4, nc = (p & 15) * 8;
            int k = 2 * kp;
#pragma unroll
            for (int e = 0; e < 8; ++e) {
                v2bf pr = {br0[i][e], br1[i][e]};
                *(v2bf*)&Bts[buf][(nc + e) * 72 + k] = pr;
            }
        }
    };

    // Prologue: fill buffer 0.
    issueA(0, 0);
    loadB(0);
    storeB(0);
    async_wait0();
    __syncthreads();

    const int nsteps = K / 64;
    for (int s = 0; s < nsteps; ++s) {
        const int cur = s & 1, nxt = cur ^ 1;
        const bool pf = (s + 1) < nsteps;
        if (pf) { issueA(nxt, (s + 1) * 64); loadB((s + 1) * 64); }

#pragma unroll
        for (int kk = 0; kk < 64; kk += 32) {
            v16bf af = load_frag(&As[cur][(w * 16) * 72 + kk], 72, lane);
#pragma unroll
            for (int j = 0; j < 8; ++j) {
                v16bf bfg = load_frag(&Bts[cur][(j * 16) * 72 + kk], 72, lane);
                acc[j] = wmma_bf16(af, bfg, acc[j]);
            }
        }

        if (pf) storeB(nxt);
        async_wait0();
        __syncthreads();
    }

    const int half = lane >> 4;
    const int nl   = lane & 15;
#pragma unroll
    for (int j = 0; j < 8; ++j) {
        int col = n0 + j * 16 + nl;
        float bv = bias[col];
#pragma unroll
        for (int c = 0; c < 8; ++c) {
            int row  = m0 + w * 16 + c + half * 8;
            float vv = acc[j][c] + bv;
            if constexpr (BF16OUT)
                ((bf16*)Cout)[(size_t)row * N + col] = (bf16)vv;
            else
                ((float*)Cout)[(size_t)row * N + col] = vv;
        }
    }
}

// ---------------------------------------------------------------------------
// Fused flash attention (forward, non-causal), log2-domain online softmax,
// double-buffered K/V tiles with async-to-LDS prefetch.
// Grid: (T/64, B*H). Block: 128 threads = 4 waves; wave w owns 16 Q rows.
// ---------------------------------------------------------------------------
__global__ __launch_bounds__(128) void flash_attn_kernel(
    const bf16* __restrict__ qkv, bf16* __restrict__ ctx) {
    __shared__ __align__(16) bf16 Kt[2][64 * 72];   // rows=key, cols=d
    __shared__ __align__(16) bf16 Vt[2][64 * 72];   // rows=d,  cols=key (V^T)
    __shared__ __align__(16) bf16 Pw[4][16 * 72];   // per-wave P staging

    const int t    = threadIdx.x;
    const int lane = t & 31;
    const int w    = t >> 5;
    const int half = lane >> 4;
    const int nl   = lane & 15;
    const int qb   = blockIdx.x;
    const int bh   = blockIdx.y;
    const int b    = bh >> 4;
    const int h    = bh & 15;

    // Q fragments from global, prescaled by (1/sqrt(dk)) * log2(e).
    const float QSCALE = 0.1803368801f;
    const bf16* qbase =
        qkv + ((size_t)(b * T_ + qb * 64 + w * 16)) * QKV_N + h * DK;
    v16bf q0 = load_frag(qbase, QKV_N, lane);
    v16bf q1 = load_frag(qbase + 32, QKV_N, lane);
#pragma unroll
    for (int e = 0; e < 16; ++e) {
        q0[e] = (bf16)((float)q0[e] * QSCALE);
        q1[e] = (bf16)((float)q1[e] * QSCALE);
    }

    v8f accO[4] = {};
    float rm[8], rl[8];
#pragma unroll
    for (int c = 0; c < 8; ++c) { rm[c] = -1.0e30f; rl[c] = 0.f; }

    auto issueK = [&](int buf, int kt) {
        const bf16* kbase =
            qkv + ((size_t)(b * T_ + kt * 64)) * QKV_N + D_MODEL + h * DK;
#pragma unroll
        for (int i = 0; i < 4; ++i) {
            int c = t + i * 128;
            int j = c >> 3, dc = (c & 7) * 8;
            async_copy_b128(&Kt[buf][j * 72 + dc], &kbase[(size_t)j * QKV_N + dc]);
        }
    };
    v8bf vr0[2], vr1[2];
    auto loadV = [&](int kt) {
        const bf16* vbase =
            qkv + ((size_t)(b * T_ + kt * 64)) * QKV_N + 2 * D_MODEL + h * DK;
#pragma unroll
        for (int i = 0; i < 2; ++i) {
            int p = t + i * 128;
            int jp = p >> 3, dc = (p & 7) * 8;
            int j = 2 * jp;
            vr0[i] = *(const v8bf*)&vbase[(size_t)j       * QKV_N + dc];
            vr1[i] = *(const v8bf*)&vbase[(size_t)(j + 1) * QKV_N + dc];
        }
    };
    auto storeV = [&](int buf) {
#pragma unroll
        for (int i = 0; i < 2; ++i) {
            int p = t + i * 128;
            int jp = p >> 3, dc = (p & 7) * 8;
            int j = 2 * jp;
#pragma unroll
            for (int e = 0; e < 8; ++e) {
                v2bf pr = {vr0[i][e], vr1[i][e]};
                *(v2bf*)&Vt[buf][(dc + e) * 72 + j] = pr;
            }
        }
    };

    // Prologue: fill buffer 0.
    issueK(0, 0);
    loadV(0);
    storeV(0);
    async_wait0();
    __syncthreads();

    const int NT = T_ / 64;
    for (int kt = 0; kt < NT; ++kt) {
        const int cur = kt & 1, nxt = cur ^ 1;
        const bool pf = (kt + 1) < NT;
        if (pf) { issueK(nxt, kt + 1); loadV(kt + 1); }

        // S = (Q*scale) * K^T : 4 key-subtiles x 2 chained WMMAs over d.
        v8f s[4];
#pragma unroll
        for (int j = 0; j < 4; ++j) {
            v8f z = {};
            v16bf k0 = load_frag(&Kt[cur][(j * 16) * 72], 72, lane);
            v16bf k1 = load_frag(&Kt[cur][(j * 16) * 72 + 32], 72, lane);
            z = wmma_bf16(q0, k0, z);
            z = wmma_bf16(q1, k1, z);
            s[j] = z;
        }

        // Online softmax in log2 domain; DPP reductions, raw v_exp_f32.
#pragma unroll
        for (int c = 0; c < 8; ++c) {
            float mx = fmaxf(fmaxf(s[0][c], s[1][c]), fmaxf(s[2][c], s[3][c]));
            mx = half_reduce_max(mx);
            float mnew = fmaxf(rm[c], mx);
            float sc   = fast_exp2(rm[c] - mnew);
            float rs   = 0.f;
#pragma unroll
            for (int j = 0; j < 4; ++j) {
                float p = fast_exp2(s[j][c] - mnew);
                s[j][c] = p;
                rs += p;
            }
            rs = half_reduce_sum(rs);
            rl[c] = rl[c] * sc + rs;
            rm[c] = mnew;
#pragma unroll
            for (int j = 0; j < 4; ++j) accO[j][c] *= sc;
        }

        // Accumulator layout -> A-fragment layout via per-wave LDS slice.
#pragma unroll
        for (int j = 0; j < 4; ++j)
#pragma unroll
            for (int c = 0; c < 8; ++c)
                Pw[w][(c + half * 8) * 72 + j * 16 + nl] = (bf16)s[j][c];

        // O += P * V
#pragma unroll
        for (int kk = 0; kk < 2; ++kk) {
            v16bf pfr = load_frag(&Pw[w][kk * 32], 72, lane);
#pragma unroll
            for (int n = 0; n < 4; ++n) {
                v16bf vf = load_frag(&Vt[cur][(n * 16) * 72 + kk * 32], 72, lane);
                accO[n] = wmma_bf16(pfr, vf, accO[n]);
            }
        }

        if (pf) storeV(nxt);
        async_wait0();
        __syncthreads();
    }

    // Normalize and store context (bf16) for the output projection GEMM.
#pragma unroll
    for (int c = 0; c < 8; ++c) {
        float inv = fast_rcp(rl[c]);
        int trow  = qb * 64 + w * 16 + c + half * 8;
#pragma unroll
        for (int n = 0; n < 4; ++n) {
            int col  = h * DK + n * 16 + nl;
            float ov = accO[n][c] * inv;
            ctx[((size_t)(b * T_) + trow) * D_MODEL + col] = (bf16)ov;
        }
    }
}

// ---------------------------------------------------------------------------
// Launcher. Inputs: x, w_qkv, b_qkv, w_out, b_out (all fp32).
// Workspace: bf16 copies of x/w_qkv/w_out, bf16 qkv, bf16 ctx (~92 MB).
// ---------------------------------------------------------------------------
extern "C" void kernel_launch(void* const* d_in, const int* in_sizes, int n_in,
                              void* d_out, int out_size, void* d_ws, size_t ws_size,
                              hipStream_t stream) {
    const float* x     = (const float*)d_in[0];
    const float* w_qkv = (const float*)d_in[1];
    const float* b_qkv = (const float*)d_in[2];
    const float* w_out = (const float*)d_in[3];
    const float* b_out = (const float*)d_in[4];

    char* ws = (char*)d_ws;
    bf16* x_bf    = (bf16*)ws;  ws += (size_t)ROWS * D_MODEL * sizeof(bf16);
    bf16* wqkv_bf = (bf16*)ws;  ws += (size_t)D_MODEL * QKV_N * sizeof(bf16);
    bf16* wout_bf = (bf16*)ws;  ws += (size_t)D_MODEL * D_MODEL * sizeof(bf16);
    bf16* qkv_bf  = (bf16*)ws;  ws += (size_t)ROWS * QKV_N * sizeof(bf16);
    bf16* ctx_bf  = (bf16*)ws;

    cvt_f32_bf16<<<1024, 256, 0, stream>>>(x, x_bf, ROWS * D_MODEL / 4);
    cvt_f32_bf16<<<512, 256, 0, stream>>>(w_qkv, wqkv_bf, D_MODEL * QKV_N / 4);
    cvt_f32_bf16<<<256, 256, 0, stream>>>(w_out, wout_bf, D_MODEL * D_MODEL / 4);

    gemm_bf16_kernel<true><<<dim3(QKV_N / 128, ROWS / 64), 128, 0, stream>>>(
        x_bf, wqkv_bf, b_qkv, qkv_bf, ROWS, QKV_N, D_MODEL);

    flash_attn_kernel<<<dim3(T_ / 64, B_ * N_HEADS), 128, 0, stream>>>(qkv_bf, ctx_bf);

    gemm_bf16_kernel<false><<<dim3(D_MODEL / 128, ROWS / 64), 128, 0, stream>>>(
        ctx_bf, wout_bf, b_out, d_out, ROWS, D_MODEL, D_MODEL);
}
